// CIN_2087354106708
// MI455X (gfx1250) — matrix operand order repaired and checked by reading, not compiled, not run on hardware
//
#include <hip/hip_runtime.h>
#include <hip/hip_bf16.h>

typedef __attribute__((ext_vector_type(16))) __bf16 v16bf;
typedef __attribute__((ext_vector_type(8)))  float  v8f;

// Problem constants
#define NB 256     // batch
#define MM 256     // M (feature dim / h dim of layer 1)
#define DD 16      // embedding dim
#define NN 4096    // NB*DD flattened "n" columns
#define H1 64
#define HNN 32

// ---------------------------------------------------------------------------
// Pack V[n][m] = x[b,m]*emb[m,d], n = b*16+d, as bf16 (n-major, m contiguous)
// ---------------------------------------------------------------------------
__global__ void k_pack_v(const float* __restrict__ x,
                         const float* __restrict__ emb,
                         __bf16* __restrict__ Vt) {
  int idx = blockIdx.x * 256 + threadIdx.x;   // < 4096*256
  int m = idx & 255;
  int n = idx >> 8;
  int b = n >> 4, d = n & 15;
  float v = x[b * MM + m] * emb[m * DD + d];
  Vt[idx] = (__bf16)v;
}

// ---------------------------------------------------------------------------
// Pack W (64 x KT*32 f32) into bf16 WMMA A-fragments.
// Fragment f = ot*KT + kt occupies 512 bf16; within: lane*16 + j, where
// element j=(2v+p) sits at K = 16*khalf + j (contiguous k order,
// matching the B fragment). Row o = ot*16 + (lane&15).
// ---------------------------------------------------------------------------
__global__ void k_pack_w(const float* __restrict__ W,
                         __bf16* __restrict__ Wp, int KT) {
  int idx = blockIdx.x * 256 + threadIdx.x;   // < 64*KT*32
  int j = idx & 15;
  int lane = (idx >> 4) & 31;
  int fidx = idx >> 9;
  int kt = fidx % KT;
  int ot = fidx / KT;
  int orow = ot * 16 + (lane & 15);
  int khalf = lane >> 4;
  int v = j >> 1, p = j & 1;
  int K = 2 * v + p + 16 * khalf;
  long Ktot = (long)KT * 32;
  Wp[idx] = (__bf16)W[(long)orow * Ktot + (long)kt * 32 + K];
}

// ---------------------------------------------------------------------------
// CIN GEMM: C[o,n] += sum_k Wrow[o,k] * Vh[h(k),n] * V[m(k),n]
// k = h*256 + m, processed in chunks of 32 (one bf16 WMMA K-step), two
// chunks per pipeline stage. Block: 256 threads = 8 waves; wave w handles
// o-half (w&1) x n-tile (w>>1). Grid: (NN/64 n-blocks, K-segments).
// Partials atomically added to C (f32).
// ---------------------------------------------------------------------------
__global__ __launch_bounds__(256) void k_gemm(
    const __bf16* __restrict__ Wp,   // packed A fragments, 4 o-tiles x KT
    const __bf16* __restrict__ Vt,   // [4096][256] bf16
    const __bf16* __restrict__ Vh,   // h-factor: [4096][hstride] bf16
    int hstride,
    float* __restrict__ C,           // [64][4096] f32 accumulation buffer
    int KT, int ktPerSeg) {
  __shared__ __align__(16) __bf16 ldsV[64][264];        // 264 = 256 + 16B pad
  __shared__ __align__(16) __bf16 ldsA[2][2][4][512];   // [buf][sub][ot]

  const int tid = threadIdx.x;
  const int nblk = blockIdx.x;
  const int ktb = blockIdx.y * ktPerSeg;

  // Cooperative load of the 64x256 bf16 V tile (rows padded in LDS)
  {
    const float4* src = (const float4*)(Vt + (size_t)nblk * 64 * 256);
    for (int i = tid; i < 64 * 32; i += 256) {
      int r = i >> 5, c = i & 31;
      *(float4*)&ldsV[r][c * 8] = src[i];
    }
  }

  const int wave = tid >> 5, lane = tid & 31;
  const int oh = wave & 1;        // which pair of o-tiles
  const int nt = wave >> 1;       // n-tile 0..3
  const int lan15 = lane & 15, khalf = lane >> 4;
  const int nlocal = nt * 16 + lan15;
  const int nglob = nblk * 64 + nlocal;
  const __bf16* vrow = &ldsV[nlocal][0];

  // A staging: 256 threads move 2 chunks x 4 fragments (8KB) per stage
  const int sOt = tid >> 6;             // 0..3
  const int sOff = (tid & 63) * 8;      // bf16 offset within fragment
  const __bf16* aBase = Wp + (size_t)sOt * KT * 512 + sOff;

  v8f acc0 = {}, acc1 = {};
  uint4 areg0 = *(const uint4*)(aBase + (size_t)(ktb + 0) * 512);
  uint4 areg1 = *(const uint4*)(aBase + (size_t)(ktb + 1) * 512);
  __bf16 vh = (__bf16)0.f;
  int hcur = -1;

  const int steps = ktPerSeg >> 1;
  for (int i = 0; i < steps; ++i) {
    const int kt0 = ktb + 2 * i;
    const int buf = i & 1;
    __syncthreads();
    *(uint4*)&ldsA[buf][0][sOt][sOff] = areg0;
    *(uint4*)&ldsA[buf][1][sOt][sOff] = areg1;
    if (i + 1 < steps) {
      areg0 = *(const uint4*)(aBase + (size_t)(kt0 + 2) * 512);
      areg1 = *(const uint4*)(aBase + (size_t)(kt0 + 3) * 512);
    }
    __syncthreads();

#pragma unroll
    for (int s = 0; s < 2; ++s) {
      const int kt = kt0 + s;
      // chunk-constant h factor (changes every 8 chunks)
      int h = kt >> 3;
      if (h != hcur) {
        hcur = h;
        vh = Vh[(size_t)nglob * hstride + h];
      }
      // Build B fragment: B[K, n] = vh * V[m0+K, n]; lanes 0-15 hold
      // K=0..15, lanes 16-31 K=16..31 (khalf); contiguous in m -> one
      // 32B LDS read; scale with packed bf16 multiplies.
      int m0 = ((kt & 7) << 5) + khalf * 16;
      v16bf pv = *(const v16bf*)&vrow[m0];
      v16bf bfrag = pv * vh;   // splat-multiply -> v_pk_mul_bf16

      v16bf a0 = *(const v16bf*)&ldsA[buf][s][oh * 2 + 0][lane * 16];
      v16bf a1 = *(const v16bf*)&ldsA[buf][s][oh * 2 + 1][lane * 16];

      acc0 = __builtin_amdgcn_wmma_f32_16x16x32_bf16(false, a0, false, bfrag,
                                                     (short)0, acc0, false, false);
      acc1 = __builtin_amdgcn_wmma_f32_16x16x32_bf16(false, a1, false, bfrag,
                                                     (short)0, acc1, false, false);
    }
  }

  // C/D layout: VGPR v, lanes 0-15 -> M=v, lanes 16-31 -> M=v+8; N = lane&15
#pragma unroll
  for (int v = 0; v < 8; ++v) {
    int o0 = (oh * 2 + 0) * 16 + v + khalf * 8;
    int o1 = (oh * 2 + 1) * 16 + v + khalf * 8;
    atomicAdd(&C[(size_t)o0 * NN + nglob], acc0[v]);
    atomicAdd(&C[(size_t)o1 * NN + nglob], acc1[v]);
  }
}

// ---------------------------------------------------------------------------
// Finish layer 1: bias + relu, sum over d; o<32 -> Vp[n][o] (bf16 factor for
// layer 2), o>=32 -> S1[b][o-32]
// ---------------------------------------------------------------------------
__global__ void k_finish1(const float* __restrict__ C1,
                          const float* __restrict__ b0,
                          float* __restrict__ S1,
                          __bf16* __restrict__ Vp) {
  int idx = blockIdx.x * 256 + threadIdx.x;  // < 64*256
  int b = idx & 255;
  int o = idx >> 8;
  float bias = b0[o];
  const float* base = C1 + (size_t)o * NN + b * 16;
  float s = 0.f, r[16];
#pragma unroll
  for (int d = 0; d < 16; ++d) {
    float t = base[d] + bias;
    t = t > 0.f ? t : 0.f;
    r[d] = t;
    s += t;
  }
  if (o < HNN) {
#pragma unroll
    for (int d = 0; d < 16; ++d)
      Vp[(size_t)(b * 16 + d) * HNN + o] = (__bf16)r[d];
  } else {
    S1[b * HNN + (o - HNN)] = s;
  }
}

// Finish layer 2: bias + relu, sum over d -> S2[b][o]
__global__ void k_finish2(const float* __restrict__ C2,
                          const float* __restrict__ b1,
                          float* __restrict__ S2) {
  int idx = blockIdx.x * 256 + threadIdx.x;  // < 64*256
  int b = idx & 255;
  int o = idx >> 8;
  float bias = b1[o];
  const float* base = C2 + (size_t)o * NN + b * 16;
  float s = 0.f;
#pragma unroll
  for (int d = 0; d < 16; ++d) {
    float t = base[d] + bias;
    s += t > 0.f ? t : 0.f;
  }
  S2[b * 64 + o] = s;
}

// Final FC: out[b, j] = fcb[j] + fcW[j, 0:32].S1[b] + fcW[j, 32:96].S2[b]
__global__ void k_final(const float* __restrict__ S1,
                        const float* __restrict__ S2,
                        const float* __restrict__ fcW,
                        const float* __restrict__ fcb,
                        float* __restrict__ out) {
  int b = threadIdx.x;  // 256 threads
  float a0 = fcb[0], a1 = fcb[1];
  for (int i = 0; i < HNN; ++i) {
    float s = S1[b * HNN + i];
    a0 += fcW[i] * s;
    a1 += fcW[96 + i] * s;
  }
  for (int i = 0; i < 64; ++i) {
    float s = S2[b * 64 + i];
    a0 += fcW[HNN + i] * s;
    a1 += fcW[96 + HNN + i] * s;
  }
  out[b * 2 + 0] = a0;
  out[b * 2 + 1] = a1;
}

extern "C" void kernel_launch(void* const* d_in, const int* in_sizes, int n_in,
                              void* d_out, int out_size, void* d_ws, size_t ws_size,
                              hipStream_t stream) {
  const float* x   = (const float*)d_in[0];
  const float* emb = (const float*)d_in[1];
  const float* W0  = (const float*)d_in[2];
  const float* b0  = (const float*)d_in[3];
  const float* W1  = (const float*)d_in[4];
  const float* b1  = (const float*)d_in[5];
  const float* fcW = (const float*)d_in[6];
  const float* fcb = (const float*)d_in[7];
  float* out = (float*)d_out;

  // Workspace carve-up (256B aligned)
  char* ws = (char*)d_ws;
  size_t off = 0;
  auto carve = [&](size_t bytes) {
    size_t o = off;
    off = (off + bytes + 255) & ~(size_t)255;
    return o;
  };
  __bf16* Vt  = (__bf16*)(ws + carve((size_t)NN * MM * 2));          // 2 MB
  __bf16* Vp  = (__bf16*)(ws + carve((size_t)NN * HNN * 2));         // 256 KB
  __bf16* W0A = (__bf16*)(ws + carve((size_t)64 * 65536 * 2));       // 8 MB
  __bf16* W1A = (__bf16*)(ws + carve((size_t)64 * 8192 * 2));        // 1 MB
  float*  C1  = (float*)(ws + carve((size_t)64 * NN * 4));           // 1 MB
  float*  C2  = (float*)(ws + carve((size_t)64 * NN * 4));           // 1 MB
  float*  S1  = (float*)(ws + carve((size_t)NB * HNN * 4));
  float*  S2  = (float*)(ws + carve((size_t)NB * 64 * 4));

  (void)hipMemsetAsync(C1, 0, (size_t)64 * NN * 4, stream);
  (void)hipMemsetAsync(C2, 0, (size_t)64 * NN * 4, stream);

  // Packing
  k_pack_v<<<(NN * MM) / 256, 256, 0, stream>>>(x, emb, Vt);
  k_pack_w<<<(64 * 2048 * 32) / 256, 256, 0, stream>>>(W0, W0A, 2048);
  k_pack_w<<<(64 * 256 * 32) / 256, 256, 0, stream>>>(W1, W1A, 256);

  // Layer 1 GEMM: K = 65536 (2048 chunks), 8 K-segments x 64 n-blocks
  k_gemm<<<dim3(NN / 64, 8), 256, 0, stream>>>(W0A, Vt, Vt, MM, C1, 2048, 256);
  k_finish1<<<(64 * NB) / 256, 256, 0, stream>>>(C1, b0, S1, Vp);

  // Layer 2 GEMM: K = 8192 (256 chunks), 4 K-segments x 64 n-blocks
  k_gemm<<<dim3(NN / 64, 4), 256, 0, stream>>>(W1A, Vt, Vp, HNN, C2, 256, 64);
  k_finish2<<<(64 * NB) / 256, 256, 0, stream>>>(C2, b1, S2);

  k_final<<<1, 256, 0, stream>>>(S1, S2, fcW, fcb, out);
}
